// LocalMambaMixing3D_13151189860536
// MI455X (gfx1250) — matrix operand order repaired and checked by reading, not compiled, not run on hardware
//
#include <hip/hip_runtime.h>
#include <hip/hip_bf16.h>

// ---------------- CDNA5 WMMA types ----------------
typedef __attribute__((ext_vector_type(16))) _Float16 v16h;
typedef __attribute__((ext_vector_type(8)))  float    v8f;

#define B_N    2
#define DIM    64
#define HID    128
#define DIN    256
#define DSTATE 16
#define DDIM   32
#define HDIM   56
#define WDIM   56
#define HW     3136          // 56*56
#define P_SP   100352        // 32*56*56
#define PT_N   6272          // P_SP / 16

// ---------------- CDNA5 async global->LDS path ----------------
#define AS1 __attribute__((address_space(1)))
#define AS3 __attribute__((address_space(3)))
typedef int v4i_gcc __attribute__((vector_size(16)));   // matches builtin's V4i param

#if __has_builtin(__builtin_amdgcn_global_load_async_to_lds_b128) && \
    __has_builtin(__builtin_amdgcn_s_wait_asynccnt)
#define USE_ASYNC_LDS 1
#else
#define USE_ASYNC_LDS 0
#endif

// copy 16 bytes global -> LDS (async when the gfx1250 builtins are available)
__device__ __forceinline__ void lds_copy16(void* lds_dst, const void* gsrc) {
#if USE_ASYNC_LDS
  __builtin_amdgcn_global_load_async_to_lds_b128((AS1 v4i_gcc*)gsrc,
                                                 (AS3 v4i_gcc*)lds_dst, 0, 0);
#else
  *(float4*)lds_dst = *(const float4*)gsrc;
#endif
}
__device__ __forceinline__ void lds_copy_wait() {
#if USE_ASYNC_LDS
  __builtin_amdgcn_s_wait_asynccnt(0);
#endif
}

// =====================================================================
// K0: one-time weight conversion W_s f32 -> f16 (0.9 MB, L2-resident)
// =====================================================================
__global__ __launch_bounds__(256) void cvt_ws_kernel(
    const float* __restrict__ W_s, _Float16* __restrict__ W_s16) {
  int i = blockIdx.x * 256 + threadIdx.x;
  if (i < 128 * 3456) W_s16[i] = (_Float16)W_s[i];
}

// =====================================================================
// K1: proj_in  xz[b,o,p] = sum_c inp[b,c,p] * W_in[o,c]   (o<256, c<64)
// M = o (16 tiles), N = p, K = c (2 WMMA steps). 1 wave = one 16x16 tile.
// =====================================================================
__global__ __launch_bounds__(256) void proj_in_kernel(
    const float* __restrict__ inp, const float* __restrict__ W_in,
    float* __restrict__ xz) {
  const int lane = threadIdx.x & 31;
  const int wv   = threadIdx.x >> 5;
  int tile = blockIdx.x * 8 + wv;          // 2*16*6272 tiles total
  int pt   = tile % PT_N;
  int rest = tile / PT_N;
  int ot   = rest & 15;
  int b    = rest >> 4;
  const int half = lane >> 4;
  const int l15  = lane & 15;
  const int m    = ot * 16 + l15;
  const int p    = pt * 16 + l15;

  const float* Arow  = W_in + (size_t)m * 64;
  const float* Bbase = inp + (size_t)b * 64 * P_SP + p;

  v8f acc = {};
  #pragma unroll
  for (int k0 = 0; k0 < 64; k0 += 32) {
    v16h a, bf;
    #pragma unroll
    for (int e = 0; e < 16; ++e) {
      int ka = k0 + 16 * half + e;                             // A 16x32 f16 layout
      a[e] = (_Float16)Arow[ka];
      int kb = k0 + 16 * half + e;                             // B 32x16 f16 layout
      bf[e] = (_Float16)Bbase[(size_t)kb * P_SP];
    }
    acc = __builtin_amdgcn_wmma_f32_16x16x32_f16(false, a, false, bf,
                                                 (short)0, acc, false, false);
  }
  float* ob = xz + (size_t)(b * 256 + ot * 16) * P_SP + (size_t)pt * 16 + l15;
  #pragma unroll
  for (int r = 0; r < 8; ++r)
    ob[(size_t)(r + 8 * half) * P_SP] = acc[r];                // D: M=r+8*half, N=l15
}

// =====================================================================
// K2: depthwise 3x3x3 conv, groups=256, SAME pad. Bandwidth-bound VALU.
// =====================================================================
__global__ __launch_bounds__(256) void dwconv_kernel(
    const float* __restrict__ xz, const float* __restrict__ W_dw,
    float* __restrict__ dw) {
  int idx = blockIdx.x;
  int d  = idx % DDIM; idx /= DDIM;
  int ch = idx % 256;
  int b  = idx / 256;
  __shared__ float wl[27];
  if (threadIdx.x < 27) wl[threadIdx.x] = W_dw[ch * 27 + threadIdx.x];
  __syncthreads();
  const float* in  = xz + (size_t)(b * 256 + ch) * P_SP;
  float*       out = dw + (size_t)(b * 256 + ch) * P_SP + (size_t)d * HW;
  for (int hw = threadIdx.x; hw < HW; hw += 256) {
    int h = hw / WDIM, w = hw % WDIM;
    float acc = 0.f;
    #pragma unroll
    for (int kd = 0; kd < 3; ++kd) {
      int dd = d + kd - 1;
      if ((unsigned)dd >= DDIM) continue;
      #pragma unroll
      for (int kh = 0; kh < 3; ++kh) {
        int hh = h + kh - 1;
        if ((unsigned)hh >= HDIM) continue;
        #pragma unroll
        for (int kw = 0; kw < 3; ++kw) {
          int ww = w + kw - 1;
          if ((unsigned)ww >= WDIM) continue;
          acc += wl[kd * 9 + kh * 3 + kw] * in[(size_t)dd * HW + hh * WDIM + ww];
        }
      }
    }
    out[hw] = acc;
  }
}

// =====================================================================
// K3: dense 3x3x3 conv (128->128) + ReLU, implicit GEMM, LDS-staged.
// Block = one p-tile x all 128 output channels (8 waves = 8 o-tiles).
//  - B tile (32k x 16n) im2col-gathered ONCE per block into LDS (f16).
//  - A slab (128 x 32 f16) staged global->LDS with async-to-LDS copies.
// Compute-bound: x1 (103MB) + W_s16 (0.9MB) fit in the 192MB L2.
// =====================================================================
#define SA_STRIDE 40   // halves; 80B row: 16B-aligned chunks + conflict-free reads
#define SB_STRIDE 36   // halves; 72B row: conflict-free fragment reads

__global__ __launch_bounds__(256) void sconv_kernel(
    const float* __restrict__ dw, const _Float16* __restrict__ W_s16,
    float* __restrict__ x1out) {
  __shared__ _Float16 sA[128 * SA_STRIDE];   // 10.0 KB
  __shared__ _Float16 sB[16 * SB_STRIDE];    // 1.1 KB
  const int tid  = threadIdx.x;
  const int lane = tid & 31;
  const int wv   = tid >> 5;                 // o-tile of this wave
  const int half = lane >> 4;
  const int l15  = lane & 15;

  int blk = blockIdx.x;                      // 2 * 6272
  int pt  = blk % PT_N;
  int b   = blk / PT_N;

  // A-slab chunk assignment: 512 x 16B chunks, 2 per thread
  const int r0 = tid >> 2, s0 = tid & 3;     // rows 0..63
  const int r1 = r0 + 64;                    // rows 64..127

  // B-element assignment: 512 elems, 2 per thread; idx -> n=idx>>5, k=idx&31
  const int nB  = tid >> 5;                  // column 0..7  (and nB+8)
  const int kB  = tid & 31;
  const int p0  = pt * 16 + nB;
  const int p1  = p0 + 8;
  const int d0 = p0 / HW, hw0 = p0 - d0 * HW, h0 = hw0 / WDIM, w0 = hw0 - h0 * WDIM;
  const int d1 = p1 / HW, hw1 = p1 - d1 * HW, h1 = hw1 / WDIM, w1 = hw1 - h1 * WDIM;

  const float* inb = dw + (size_t)b * 256 * P_SP;   // x1 half = channels 0..127
  const int m = wv * 16 + l15;                      // A row for this lane
  const int p = pt * 16 + l15;

  v8f acc = {};
  for (int k0 = 0; k0 < 3456; k0 += 32) {
    __syncthreads();   // previous iteration's LDS reads complete

    // ---- stage A slab via CDNA5 async global->LDS ----
    lds_copy16(&sA[r0 * SA_STRIDE + s0 * 8], W_s16 + (size_t)r0 * 3456 + k0 + s0 * 8);
    lds_copy16(&sA[r1 * SA_STRIDE + s0 * 8], W_s16 + (size_t)r1 * 3456 + k0 + s0 * 8);

    // ---- build B tile (im2col gather, once per block) ----
    {
      int kk = k0 + kB;
      int c  = kk / 27;  int t  = kk - c * 27;
      int kd = t / 9;    int r9 = t - kd * 9;
      int kh = r9 / 3;   int kw = r9 - kh * 3;
      const float* cp = inb + (size_t)c * P_SP;
      int dd = d0 + kd - 1, hh = h0 + kh - 1, ww = w0 + kw - 1;
      float v0 = 0.f;
      if ((unsigned)dd < DDIM && (unsigned)hh < HDIM && (unsigned)ww < WDIM)
        v0 = cp[(size_t)dd * HW + hh * WDIM + ww];
      sB[nB * SB_STRIDE + kB] = (_Float16)v0;
      dd = d1 + kd - 1; hh = h1 + kh - 1; ww = w1 + kw - 1;
      float v1 = 0.f;
      if ((unsigned)dd < DDIM && (unsigned)hh < HDIM && (unsigned)ww < WDIM)
        v1 = cp[(size_t)dd * HW + hh * WDIM + ww];
      sB[(nB + 8) * SB_STRIDE + kB] = (_Float16)v1;
    }
    lds_copy_wait();    // drain this wave's ASYNCcnt
    __syncthreads();    // publish LDS to all 8 waves

    // ---- fragments from LDS + WMMA ----
    v16h a, bf;
    #pragma unroll
    for (int e = 0; e < 16; ++e) {
      int ka = 16 * half + e;
      a[e]  = sA[m * SA_STRIDE + ka];
      bf[e] = sB[l15 * SB_STRIDE + 16 * half + e];
    }
    acc = __builtin_amdgcn_wmma_f32_16x16x32_f16(false, a, false, bf,
                                                 (short)0, acc, false, false);
  }

  float* ob = x1out + (size_t)(b * 128 + wv * 16) * P_SP + p;
  #pragma unroll
  for (int r = 0; r < 8; ++r) {
    float v = acc[r];
    ob[(size_t)(r + 8 * half) * P_SP] = v > 0.f ? v : 0.f;     // fused ReLU
  }
}

// =====================================================================
// K4: global HxW mean pool of x2 half. block -> (b, c, d).
// =====================================================================
__global__ __launch_bounds__(256) void pool_kernel(
    const float* __restrict__ dw, float* __restrict__ pooled) {
  int idx = blockIdx.x;
  int d = idx % DDIM; idx /= DDIM;
  int c = idx % HID;
  int b = idx / HID;
  const float* in = dw + (size_t)(b * 256 + HID + c) * P_SP + (size_t)d * HW;
  float s = 0.f;
  for (int i = threadIdx.x; i < HW; i += 256) s += in[i];
  __shared__ float red[256];
  red[threadIdx.x] = s;
  __syncthreads();
  for (int off = 128; off > 0; off >>= 1) {
    if (threadIdx.x < off) red[threadIdx.x] += red[threadIdx.x + off];
    __syncthreads();
  }
  if (threadIdx.x == 0)
    pooled[(size_t)(b * HID + c) * DDIM + d] = red[0] * (1.f / HW);
}

// =====================================================================
// K5: full temporal branch: W_t1 -> Mamba (L=32) -> W_t2 -> sigmoid.
// One workgroup per batch; one thread per DIN channel; sequence in LDS.
// =====================================================================
__global__ __launch_bounds__(256) void temporal_kernel(
    const float* __restrict__ pooled, const float* __restrict__ W_t1,
    const float* __restrict__ W_t2,  const float* __restrict__ m_in_w,
    const float* __restrict__ m_conv_w, const float* __restrict__ m_conv_b,
    const float* __restrict__ m_x_w, const float* __restrict__ m_dt_w,
    const float* __restrict__ m_dt_b, const float* __restrict__ m_A_log,
    const float* __restrict__ m_D, const float* __restrict__ m_out_w,
    float* __restrict__ g) {
  const int b   = blockIdx.x;
  const int tid = threadIdx.x;
  __shared__ float s_seq[32 * 128];
  __shared__ float s_x[32 * 256];
  __shared__ float s_dtr[32 * 8];
  __shared__ float s_B[32 * 16];
  __shared__ float s_C[32 * 16];

  for (int idx = tid; idx < 32 * 128; idx += 256) {
    int l = idx >> 7, o = idx & 127;
    const float* pr = pooled + (size_t)b * 128 * 32 + l;
    const float* wr = W_t1 + o * 128;
    float s = 0.f;
    for (int c = 0; c < 128; ++c) s += pr[c * 32] * wr[c];
    s_seq[idx] = s;
  }
  __syncthreads();

  const int dch = tid;
  float xr[32], zr[32];
  {
    const float* wx = m_in_w + (size_t)dch * 128;
    const float* wz = m_in_w + (size_t)(256 + dch) * 128;
    for (int l = 0; l < 32; ++l) {
      const float* sr = s_seq + l * 128;
      float sx = 0.f, sz = 0.f;
      for (int c = 0; c < 128; ++c) { float v = sr[c]; sx += v * wx[c]; sz += v * wz[c]; }
      xr[l] = sx; zr[l] = sz;
    }
  }
  {
    float c0 = m_conv_w[dch * 4 + 0], c1 = m_conv_w[dch * 4 + 1];
    float c2 = m_conv_w[dch * 4 + 2], c3 = m_conv_w[dch * 4 + 3];
    float cb = m_conv_b[dch];
    float xm3 = 0.f, xm2 = 0.f, xm1 = 0.f;
    for (int l = 0; l < 32; ++l) {
      float x0 = xr[l];
      float v  = cb + c0 * xm3 + c1 * xm2 + c2 * xm1 + c3 * x0;
      xm3 = xm2; xm2 = xm1; xm1 = x0;
      float sv = v / (1.f + __expf(-v));
      xr[l] = sv;
      s_x[l * 256 + dch] = sv;
    }
  }
  __syncthreads();

  for (int idx = tid; idx < 32 * 40; idx += 256) {
    int l = idx / 40, j = idx - l * 40;
    const float* xrow = s_x + l * 256;
    const float* wr   = m_x_w + j * 256;
    float s = 0.f;
    for (int c = 0; c < 256; ++c) s += xrow[c] * wr[c];
    if (j < 8)       s_dtr[l * 8 + j] = s;
    else if (j < 24) s_B[l * 16 + (j - 8)] = s;
    else             s_C[l * 16 + (j - 24)] = s;
  }
  __syncthreads();

  {
    float wdt[8];
    #pragma unroll
    for (int r = 0; r < 8; ++r) wdt[r] = m_dt_w[dch * 8 + r];
    float dtb = m_dt_b[dch];
    float Arow[DSTATE];
    #pragma unroll
    for (int n = 0; n < DSTATE; ++n) Arow[n] = -__expf(m_A_log[dch * DSTATE + n]);
    float Dp = m_D[dch];
    float h[DSTATE];
    #pragma unroll
    for (int n = 0; n < DSTATE; ++n) h[n] = 0.f;
    for (int l = 0; l < 32; ++l) {
      float dtv = dtb;
      #pragma unroll
      for (int r = 0; r < 8; ++r) dtv += s_dtr[l * 8 + r] * wdt[r];
      float dt = (dtv > 20.f) ? dtv : __logf(1.f + __expf(dtv));   // softplus
      float xv = xr[l];
      float y = 0.f;
      #pragma unroll
      for (int n = 0; n < DSTATE; ++n) {
        float dA = __expf(dt * Arow[n]);
        float hb = dA * h[n] + dt * s_B[l * 16 + n] * xv;
        h[n] = hb;
        y += hb * s_C[l * 16 + n];
      }
      y += xv * Dp;
      float zv = zr[l];
      y *= zv / (1.f + __expf(-zv));     // * silu(z)
      xr[l] = y;
    }
  }
  __syncthreads();
  for (int l = 0; l < 32; ++l) s_x[l * 256 + dch] = xr[l];   // s_x now holds y
  __syncthreads();

  for (int idx = tid; idx < 32 * 128; idx += 256) {
    int l = idx >> 7, c = idx & 127;
    const float* yrow = s_x + l * 256;
    const float* wr   = m_out_w + c * 256;
    float s = 0.f;
    for (int dd = 0; dd < 256; ++dd) s += yrow[dd] * wr[dd];
    s_seq[idx] = s;
  }
  __syncthreads();

  for (int idx = tid; idx < 128 * 32; idx += 256) {
    int ch = idx >> 5, l = idx & 31;
    const float* wr = W_t2 + ch * 128;
    float s = 0.f;
    for (int c = 0; c < 128; ++c) s += s_seq[l * 128 + c] * wr[c];
    g[(size_t)b * 128 * 32 + ch * 32 + l] = 1.f / (1.f + __expf(-s));
  }
}

// =====================================================================
// K6: out[b,oc,p] = sum_c W_out[oc,c] * (x1[b,c,p] * g[b,c,d(p)])
// M = oc (4 tiles), N = p, K = c (4 WMMA steps). Gate fused into B frag.
// =====================================================================
__global__ __launch_bounds__(256) void out_kernel(
    const float* __restrict__ x1, const float* __restrict__ g,
    const float* __restrict__ W_out, float* __restrict__ out) {
  const int lane = threadIdx.x & 31;
  const int wv   = threadIdx.x >> 5;
  int tile = blockIdx.x * 8 + wv;          // 2*4*6272 tiles
  int pt   = tile % PT_N;
  int rest = tile / PT_N;
  int ot   = rest & 3;
  int b    = rest >> 2;
  const int half = lane >> 4;
  const int l15  = lane & 15;
  const int m    = ot * 16 + l15;
  const int p    = pt * 16 + l15;
  const int d    = p / HW;

  const float* Arow = W_out + (size_t)m * 128;
  const float* xb   = x1 + (size_t)b * 128 * P_SP + p;
  const float* gb   = g + (size_t)b * 128 * 32 + d;

  v8f acc = {};
  #pragma unroll
  for (int k0 = 0; k0 < 128; k0 += 32) {
    v16h a, bf;
    #pragma unroll
    for (int e = 0; e < 16; ++e) {
      int ka = k0 + 16 * half + e;
      a[e] = (_Float16)Arow[ka];
      int kb = k0 + 16 * half + e;
      bf[e] = (_Float16)(xb[(size_t)kb * P_SP] * gb[(size_t)kb * 32]);
    }
    acc = __builtin_amdgcn_wmma_f32_16x16x32_f16(false, a, false, bf,
                                                 (short)0, acc, false, false);
  }
  float* ob = out + (size_t)(b * 64 + ot * 16) * P_SP + p;
  #pragma unroll
  for (int r = 0; r < 8; ++r)
    ob[(size_t)(r + 8 * half) * P_SP] = acc[r];
}

// =====================================================================
extern "C" void kernel_launch(void* const* d_in, const int* in_sizes, int n_in,
                              void* d_out, int out_size, void* d_ws, size_t ws_size,
                              hipStream_t stream) {
  (void)in_sizes; (void)n_in; (void)out_size; (void)ws_size;
  const float* inp      = (const float*)d_in[0];
  const float* W_in     = (const float*)d_in[1];
  const float* W_dw     = (const float*)d_in[2];
  const float* W_s      = (const float*)d_in[3];
  const float* W_t1     = (const float*)d_in[4];
  const float* W_t2     = (const float*)d_in[5];
  const float* m_in_w   = (const float*)d_in[6];
  const float* m_conv_w = (const float*)d_in[7];
  const float* m_conv_b = (const float*)d_in[8];
  const float* m_x_w    = (const float*)d_in[9];
  const float* m_dt_w   = (const float*)d_in[10];
  const float* m_dt_b   = (const float*)d_in[11];
  const float* m_A_log  = (const float*)d_in[12];
  const float* m_D      = (const float*)d_in[13];
  const float* m_out_w  = (const float*)d_in[14];
  const float* W_out    = (const float*)d_in[15];
  float* out = (float*)d_out;

  // workspace layout (floats): xz region is reused for the conv output
  float* ws = (float*)d_ws;
  const size_t SZ = (size_t)2 * 256 * P_SP;     // 51,380,224 floats
  float* xz       = ws;                          // K1 out; reused as x1 conv out
  float* dwbuf    = ws + SZ;                     // K2 out
  float* pooled   = ws + 2 * SZ;                 // 8192 floats
  float* gbuf     = pooled + 8192;               // 8192 floats
  _Float16* Ws16  = (_Float16*)(gbuf + 8192);    // 442,368 halves (16B-aligned)

  cvt_ws_kernel  <<<1728,  256, 0, stream>>>(W_s, Ws16);
  proj_in_kernel <<<25088, 256, 0, stream>>>(inp, W_in, xz);
  dwconv_kernel  <<<16384, 256, 0, stream>>>(xz, W_dw, dwbuf);
  sconv_kernel   <<<12544, 256, 0, stream>>>(dwbuf, Ws16, xz);  // xz := relu(conv(x1))
  pool_kernel    <<<8192,  256, 0, stream>>>(dwbuf, pooled);
  temporal_kernel<<<2,     256, 0, stream>>>(pooled, W_t1, W_t2, m_in_w, m_conv_w,
                                             m_conv_b, m_x_w, m_dt_w, m_dt_b,
                                             m_A_log, m_D, m_out_w, gbuf);
  out_kernel     <<<6272,  256, 0, stream>>>(xz, gbuf, W_out, out);
}
